// Encoder_88321707475257
// MI455X (gfx1250) — matrix operand order repaired and checked by reading, not compiled, not run on hardware
//
#include <hip/hip_runtime.h>
#include <hip/hip_bf16.h>

// ---------------------------------------------------------------------------
// Bidirectional LSTM encoder for MI455X (gfx1250, wave32, WMMA bf16->f32).
// B=32, T=512, V=32000, E=256, U=512, 4U=2048.
//
//   1) convert emb -> bf16; transpose W/U into column-major bf16.
//   2) input projection GEMM (WMMA): xz[t][b][g] = emb[x[b,t]] @ W + bias.
//   3) persistent recurrent scan: 32 WGs (16 per direction). Each WG stages
//      its 128KB recurrent-weight slice into LDS once and the 32KB h buffer
//      into LDS each step via CDNA5 async global->LDS (ASYNCcnt); the WMMA
//      inner loop runs entirely out of LDS. Per-direction grid barrier:
//      hardware cluster barrier when launched in cluster mode, L2-atomic
//      fallback otherwise.
// ---------------------------------------------------------------------------

typedef __bf16 bf16;
typedef __attribute__((ext_vector_type(16))) __bf16 v16bf;
typedef __attribute__((ext_vector_type(8)))  float  v8f;
typedef unsigned int u32;

#define B_  32
#define T_  512
#define V_  32000
#define E_  256
#define U_  512
#define G4  2048   // 4*U

// LDS layout for the scan kernel (dynamic shared memory)
#define SM_W   0                 // 128 cols x 512 halfs bf16 = 131072 B
#define SM_H   131072            // 32 x 512 halfs bf16       =  32768 B
#define SM_Z   163840            // 4 x 32 x 32 f32           =  16384 B
#define SM_TOT 180224

// ---- fragment loaders -----------------------------------------------------
// A-matrix 16x32 bf16: per lane two 8-half chunks at
// k0=(lane>=16?16:0) and k0+8.  B-matrix 32x16: per lane one contiguous
// 16-half chunk of its column at k0=(lane>=16?16:0).
static __device__ __forceinline__ v16bf ld_frag2(const bf16* p0, const bf16* p1) {
    union { v16bf v; uint4 q[2]; } u;
    u.q[0] = *reinterpret_cast<const uint4*>(p0);
    u.q[1] = *reinterpret_cast<const uint4*>(p1);
    return u.v;
}
static __device__ __forceinline__ v16bf ld_frag1(const bf16* p) {
    union { v16bf v; uint4 q[2]; } u;
    u.q[0] = *reinterpret_cast<const uint4*>(p);
    u.q[1] = *reinterpret_cast<const uint4*>(p + 8);
    return u.v;
}
static __device__ __forceinline__ v8f wmma_bf16(v16bf a, v16bf b, v8f c) {
    return __builtin_amdgcn_wmma_f32_16x16x32_bf16(false, a, false, b,
                                                   (short)0, c, false, false);
}
static __device__ __forceinline__ float sigmoidf_(float x) {
    return 1.0f / (1.0f + __expf(-x));
}

// ---- CDNA5 async global->LDS (ASYNCcnt) -----------------------------------
// Per-lane: copies 16 bytes MEM[saddr + voff] -> LDS[lds_addr].
static __device__ __forceinline__ void async_g2l_b128(u32 lds_addr,
                                                      const void* sbase,
                                                      u32 voff) {
    asm volatile("global_load_async_to_lds_b128 %0, %1, %2"
                 :: "v"(lds_addr), "v"(voff), "s"(sbase)
                 : "memory");
}
static __device__ __forceinline__ void wait_asynccnt0() {
    asm volatile("s_wait_asynccnt 0x0" ::: "memory");
}
static __device__ __forceinline__ u32 lds_off_of(const void* p) {
    // generic shared pointer: low 32 bits are the wave-relative LDS offset
    return (u32)(uintptr_t)p;
}

// ---- prep kernels ---------------------------------------------------------
__global__ void cvt_bf16_kernel(const float* __restrict__ src,
                                bf16* __restrict__ dst, int n) {
    int i = blockIdx.x * 256 + threadIdx.x;
    if (i < n) dst[i] = (bf16)src[i];
}

// src[K][N] f32 row-major -> dst[N][K] bf16 (column-contiguous over K)
__global__ void transpose_bf16_kernel(const float* __restrict__ src,
                                      bf16* __restrict__ dst, int K, int N) {
    int i = blockIdx.x * 256 + threadIdx.x;
    if (i < K * N) {
        int k = i / N, n = i % N;
        dst[n * K + k] = (bf16)src[i];
    }
}

// init h double-buffer phase 0 from h0_f / h0_b; zero the grid barriers.
__global__ void hinit_kernel(const float* __restrict__ h0F,
                             const float* __restrict__ h0B,
                             bf16* __restrict__ hbuf,
                             u32* __restrict__ bar) {
    int i = blockIdx.x * 256 + threadIdx.x;
    if (i < 4) bar[i] = 0u;
    if (i < 2 * B_ * U_) {
        int dir = i >> 14, r = i & 16383;                 // 16384 = B_*U_
        hbuf[(0 * 2 + dir) * (B_ * U_) + r] = (bf16)(dir ? h0B[r] : h0F[r]);
    }
}

// ---- input projection: xz[t*32+b][n] = emb_bf16[x[b,t]] @ Wt + bias -------
// grid (32, 512, 2), block 128 (4 waves). wave -> one N-tile of 16 columns,
// both M-tiles (rows m0..m0+31 => t = blockIdx.y, all 32 batches).
__global__ __launch_bounds__(128) void proj_kernel(
    const int*  __restrict__ xidx,   // [B,T]
    const bf16* __restrict__ embBF,  // [V,E]
    const bf16* __restrict__ WtF,    // [G4,E] (transposed)
    const bf16* __restrict__ WtB,
    const float* __restrict__ bF, const float* __restrict__ bB,
    float* __restrict__ xzF, float* __restrict__ xzB)
{
    const int dir   = blockIdx.z;
    const int t     = blockIdx.y;                 // 0..511
    const int wave  = threadIdx.x >> 5;
    const int lane  = threadIdx.x & 31;
    const int ntile = blockIdx.x * 4 + wave;      // 0..127
    const int laneN = lane & 15;
    const int hi8   = (lane >> 4) * 8;
    const int k0a   = (lane >> 4) * 16;           // A chunk base (halves)
    const int k0b   = (lane >> 4) * 16;           // B chunk base (halves)

    const bf16*  Wt   = dir ? WtB : WtF;
    const float* bias = dir ? bB : bF;
    float*       xz   = dir ? xzB : xzF;

    const int n = ntile * 16 + laneN;
    const bf16* rowA = embBF + (size_t)xidx[laneN * T_ + t] * E_;        // b = laneN
    const bf16* rowB = embBF + (size_t)xidx[(16 + laneN) * T_ + t] * E_; // b = 16+laneN
    const bf16* wrow = Wt + (size_t)n * E_;

    v8f acc0 = {}, acc1 = {};
#pragma unroll
    for (int k = 0; k < E_ / 32; ++k) {
        v16bf A0 = ld_frag2(rowA + k * 32 + k0a, rowA + k * 32 + k0a + 8);
        v16bf A1 = ld_frag2(rowB + k * 32 + k0a, rowB + k * 32 + k0a + 8);
        v16bf Bf = ld_frag1(wrow + k * 32 + k0b);
        acc0 = wmma_bf16(A0, Bf, acc0);
        acc1 = wmma_bf16(A1, Bf, acc1);
    }
    const float bv = bias[n];
    float* baseout = xz + ((size_t)t * B_) * G4 + n;   // row m = t*32+b
#pragma unroll
    for (int r = 0; r < 8; ++r) {
        baseout[(size_t)(r + hi8) * G4]      = acc0[r] + bv;
        baseout[(size_t)(16 + r + hi8) * G4] = acc1[r] + bv;
    }
}

// ---- per-direction grid barrier -------------------------------------------
// Cluster fast path: the 16 workgroups of one direction form exactly one
// CDNA5 workgroup-cluster; s_barrier_signal/wait -3 is a hardware barrier.
// Fallback: device-scope atomic counter in L2.
static __device__ __forceinline__ void sync_dir(u32* cnt, u32* gen,
                                                u32 target, bool use_cluster) {
    __syncthreads();
    if (use_cluster) {
        __threadfence();                       // release h stores
        __builtin_amdgcn_s_cluster_barrier();  // s_barrier_signal/wait -3
        __threadfence();                       // acquire
    } else {
        if (threadIdx.x == 0) {
            __threadfence();
            u32 prev = __hip_atomic_fetch_add(cnt, 1u, __ATOMIC_ACQ_REL,
                                              __HIP_MEMORY_SCOPE_AGENT);
            if (prev == 15u) {
                __hip_atomic_store(cnt, 0u, __ATOMIC_RELAXED,
                                   __HIP_MEMORY_SCOPE_AGENT);
                __hip_atomic_fetch_add(gen, 1u, __ATOMIC_ACQ_REL,
                                       __HIP_MEMORY_SCOPE_AGENT);
            } else {
                while (__hip_atomic_load(gen, __ATOMIC_ACQUIRE,
                                         __HIP_MEMORY_SCOPE_AGENT) < target) {
                    __builtin_amdgcn_s_sleep(8);
                }
            }
        }
        __syncthreads();
        __threadfence();                       // acquire for all lanes
    }
}

// ---- persistent bidirectional LSTM scan -----------------------------------
// grid 32 (16 u-tiles x 2 dirs), block 256 (8 waves).
// wave w: gate = w>>1, 16 columns at ubase + (w&1)*16.
__global__ __launch_bounds__(256) void lstm_scan_kernel(
    const float* __restrict__ xzF, const float* __restrict__ xzB,
    const bf16* __restrict__ UtF,  const bf16* __restrict__ UtB,  // [G4,U]
    const float* __restrict__ c0F, const float* __restrict__ c0B,
    bf16* __restrict__ hbuf,       // [2 phase][2 dir][B][U] bf16
    float* __restrict__ out,       // [B,T,2U] then [B,2U]
    u32* __restrict__ bar)         // [dir][cnt,gen]
{
    extern __shared__ char smem[];
    bf16*  ldsW = (bf16*) (smem + SM_W);   // [128][U_] local weight slice
    bf16*  ldsH = (bf16*) (smem + SM_H);   // [B_][U_] h broadcast
    float* zbuf = (float*)(smem + SM_Z);   // [4][B_][32] gate pre-activations

    const int wgid  = blockIdx.x;        // 0..31
    const int dir   = wgid >> 4;
    const int utile = wgid & 15;
    const int ubase = utile * 32;
    const int tid   = threadIdx.x;
    const int wave  = tid >> 5;
    const int lane  = tid & 31;
    const int gate  = wave >> 1;
    const int nsub  = (wave & 1) * 16;
    const int laneN = lane & 15;
    const int hi8   = (lane >> 4) * 8;
    const int k0a   = (lane >> 4) * 16;
    const int k0b   = (lane >> 4) * 16;

    const float* xz = dir ? xzB : xzF;
    const bf16*  Ut = dir ? UtB : UtF;
    const float* c0 = dir ? c0B : c0F;
    u32* cnt = bar + dir * 2;
    u32* gen = bar + dir * 2 + 1;

    // In-cluster? IB_STS2.Cluster_ID[9:6] != 0  (hwreg id 28, offset 6, sz 4)
    const bool use_cluster =
        (__builtin_amdgcn_s_getreg((3 << 11) | (6 << 6) | 28) != 0);

    // -- stage this WG's recurrent weight slice into LDS (async, once) --
    // local row lc (0..127): gate g=lc>>5, unit wi=lc&31
    //   <- global row g*512 + ubase + wi of Ut[dir]  (row = 1024 bytes)
    {
        const u32 ldsWb = lds_off_of(ldsW);
#pragma unroll
        for (int i = 0; i < 32; ++i) {
            int flat = tid + i * 256;            // 0..8191 16B-chunks
            int lc   = flat >> 6;                // 64 chunks per row
            int c16  = flat & 63;
            int g    = lc >> 5, wi = lc & 31;
            u32 voff  = (u32)((g * U_ + ubase + wi) * (U_ * 2) + c16 * 16);
            u32 laddr = ldsWb + (u32)(lc * (U_ * 2) + c16 * 16);
            async_g2l_b128(laddr, Ut, voff);
        }
        wait_asynccnt0();
    }
    __syncthreads();

    // -- c-state: thread owns 4 (b,u) elements --
    float c[4];
#pragma unroll
    for (int j = 0; j < 4; ++j) {
        int e = tid + j * 256, b = e >> 5, uu = e & 31;
        c[j] = c0[b * U_ + ubase + uu];
    }

    const int lcol = gate * 32 + nsub + laneN;   // local weight row (0..127)
    const u32 ldsHb = lds_off_of(ldsH);

    for (int s = 0; s < T_; ++s) {
        const int t = dir ? (T_ - 1 - s) : s;
        const bf16* hb = hbuf + ((s & 1) * 2 + dir) * (B_ * U_);
        const int ncol = gate * U_ + ubase + nsub + laneN;

        // async-copy h (32KB) into LDS; overlap with xz seed loads
#pragma unroll
        for (int i = 0; i < 8; ++i) {
            u32 off = (u32)((tid + i * 256) * 16);
            async_g2l_b128(ldsHb + off, hb, off);
        }

        // accumulators seeded with xz[t]  (z = xz + h @ U)
        v8f acc0, acc1;
        {
            const float* base = xz + ((size_t)t * B_) * G4 + ncol;
#pragma unroll
            for (int r = 0; r < 8; ++r) {
                acc0[r] = base[(size_t)(r + hi8) * G4];
                acc1[r] = base[(size_t)(16 + r + hi8) * G4];
            }
        }
        wait_asynccnt0();
        __syncthreads();

        // h @ U entirely out of LDS
#pragma unroll
        for (int k = 0; k < 16; ++k) {
            const bf16* ha = ldsH + laneN * U_ + k * 32 + k0a;   // b = 0..15
            v16bf A0 = ld_frag2(ha, ha + 8);
            v16bf A1 = ld_frag2(ha + 16 * U_, ha + 16 * U_ + 8); // b = 16..31
            v16bf Bf = ld_frag1(ldsW + (size_t)lcol * U_ + k * 32 + k0b);
            acc0 = wmma_bf16(A0, Bf, acc0);
            acc1 = wmma_bf16(A1, Bf, acc1);
        }

        // scatter z tiles to LDS for cross-gate fusion
#pragma unroll
        for (int r = 0; r < 8; ++r) {
            zbuf[(gate * B_ + (r + hi8))      * 32 + nsub + laneN] = acc0[r];
            zbuf[(gate * B_ + (16 + r + hi8)) * 32 + nsub + laneN] = acc1[r];
        }
        __syncthreads();

        // fused gate math; write h to next phase buffer + outputs
        bf16* hnext = hbuf + (((s + 1) & 1) * 2 + dir) * (B_ * U_);
#pragma unroll
        for (int j = 0; j < 4; ++j) {
            int e = tid + j * 256, b = e >> 5, uu = e & 31;
            float zi = zbuf[(0 * B_ + b) * 32 + uu];
            float zf = zbuf[(1 * B_ + b) * 32 + uu];
            float zg = zbuf[(2 * B_ + b) * 32 + uu];
            float zo = zbuf[(3 * B_ + b) * 32 + uu];
            c[j] = sigmoidf_(zf) * c[j] + sigmoidf_(zi) * tanhf(zg);
            float h = sigmoidf_(zo) * tanhf(c[j]);
            int u = ubase + uu;
            hnext[b * U_ + u] = (bf16)h;
            out[((size_t)b * T_ + t) * (2 * U_) + dir * U_ + u] = h;
            if (s == T_ - 1)
                out[(size_t)B_ * T_ * (2 * U_) + b * (2 * U_) + dir * U_ + u] = h;
        }
        // per-direction barrier (covers zbuf/ldsH reuse: starts with syncthreads)
        sync_dir(cnt, gen, (u32)(s + 1), use_cluster);
    }
}

// ---- workspace layout (bytes) ---------------------------------------------
static constexpr size_t OFF_XZF  = 0;
static constexpr size_t OFF_XZB  = OFF_XZF + (size_t)T_*B_*G4*4;
static constexpr size_t OFF_EMB  = OFF_XZB + (size_t)T_*B_*G4*4;
static constexpr size_t OFF_WTF  = OFF_EMB + (size_t)V_*E_*2;
static constexpr size_t OFF_WTB  = OFF_WTF + (size_t)G4*E_*2;
static constexpr size_t OFF_UTF  = OFF_WTB + (size_t)G4*E_*2;
static constexpr size_t OFF_UTB  = OFF_UTF + (size_t)G4*U_*2;
static constexpr size_t OFF_HBUF = OFF_UTB + (size_t)G4*U_*2;
static constexpr size_t OFF_BAR  = OFF_HBUF + (size_t)2*2*B_*U_*2;

extern "C" void kernel_launch(void* const* d_in, const int* in_sizes, int n_in,
                              void* d_out, int out_size, void* d_ws, size_t ws_size,
                              hipStream_t stream) {
    (void)in_sizes; (void)n_in; (void)out_size; (void)ws_size;
    const int*   x    = (const int*)  d_in[0];
    const float* h0_f = (const float*)d_in[1];
    const float* c0_f = (const float*)d_in[2];
    const float* h0_b = (const float*)d_in[3];
    const float* c0_b = (const float*)d_in[4];
    const float* emb  = (const float*)d_in[5];
    const float* W_f  = (const float*)d_in[6];
    const float* U_f  = (const float*)d_in[7];
    const float* b_f  = (const float*)d_in[8];
    const float* W_b  = (const float*)d_in[9];
    const float* U_b  = (const float*)d_in[10];
    const float* b_b  = (const float*)d_in[11];
    float* out = (float*)d_out;

    char* ws = (char*)d_ws;
    float* xzF  = (float*)(ws + OFF_XZF);
    float* xzB  = (float*)(ws + OFF_XZB);
    bf16*  embB = (bf16*)(ws + OFF_EMB);
    bf16*  WtF  = (bf16*)(ws + OFF_WTF);
    bf16*  WtB  = (bf16*)(ws + OFF_WTB);
    bf16*  UtF  = (bf16*)(ws + OFF_UTF);
    bf16*  UtB  = (bf16*)(ws + OFF_UTB);
    bf16*  hbuf = (bf16*)(ws + OFF_HBUF);
    u32*   bar  = (u32*) (ws + OFF_BAR);

    // 1) conversions / transposes / init
    cvt_bf16_kernel<<<(V_*E_ + 255)/256, 256, 0, stream>>>(emb, embB, V_*E_);
    transpose_bf16_kernel<<<(E_*G4 + 255)/256, 256, 0, stream>>>(W_f, WtF, E_, G4);
    transpose_bf16_kernel<<<(E_*G4 + 255)/256, 256, 0, stream>>>(W_b, WtB, E_, G4);
    transpose_bf16_kernel<<<(U_*G4 + 255)/256, 256, 0, stream>>>(U_f, UtF, U_, G4);
    transpose_bf16_kernel<<<(U_*G4 + 255)/256, 256, 0, stream>>>(U_b, UtB, U_, G4);
    hinit_kernel<<<(2*B_*U_ + 255)/256, 256, 0, stream>>>(h0_f, h0_b, hbuf, bar);

    // 2) input projection (both directions), WMMA bf16
    proj_kernel<<<dim3(G4/64, T_, 2), 128, 0, stream>>>(
        x, embB, WtF, WtB, b_f, b_b, xzF, xzB);

    // 3) persistent bidirectional scan: WMMA + async->LDS + cluster barrier
    lstm_scan_kernel<<<32, 256, SM_TOT, stream>>>(
        xzF, xzB, UtF, UtB, c0_f, c0_b, hbuf, out, bar);
}
